// ExpertBranch_15496242004074
// MI455X (gfx1250) — compile-verified, matrix operand order checked by reading, not run on hardware
//
#include <hip/hip_runtime.h>
#include <hip/hip_bf16.h>

// MI455X (gfx1250) fp8 block-scale GEMM pipeline:
//   quantize(x) -> fp8 + per-(row,128-block) scale
//   quantize+pack(w) -> WMMA-B-layout fp8 + per-(128-block,col) scale
//   GEMM via V_WMMA_F32_16X16X128_FP8_FP8, one WMMA per quant block,
//   B fragments double-buffered in LDS via GLOBAL_LOAD_ASYNC_TO_LDS_B128,
//   WMMAs issued in independent pairs to hide the WMMA->VALU hazard,
//   partials scaled by sx*sw and FMA'd into f32 accumulators.

typedef int   v16i __attribute__((ext_vector_type(16)));
typedef int   v4i  __attribute__((ext_vector_type(4)));
typedef float v8f  __attribute__((ext_vector_type(8)));

#define FP8_MAX_F 448.0f
#define QEPS      1e-12f

// Software f32 -> e4m3 encode: clamp to 448, round-to-nearest-even.
__device__ __forceinline__ unsigned int f32_to_fp8_e4m3(float f) {
    unsigned int s = (__float_as_uint(f) >> 24) & 0x80u;
    float a = fabsf(f);
    a = fminf(a, 448.0f);
    unsigned int r;
    if (a < 0.015625f) {                          // below 2^-6: denormal range
        r = (unsigned int)(int)rintf(a * 512.0f); // 0..8 (8 == min normal)
    } else {
        unsigned int u = __float_as_uint(a);
        u += 0x0007FFFFu + ((u >> 20) & 1u);      // RNE keep 3 mantissa bits
        unsigned int e = ((u >> 23) & 0xFFu) - 120u; // -127 + 7 bias
        r = (e << 3) | ((u >> 20) & 7u);
    }
    return s | r;
}

// One wave (32 lanes) quantizes one 128-element block along the last axis.
// Writes fp8 row-major q[M,K] and transposed scales st[KB, M].
__global__ __launch_bounds__(256)
void quant_rows_kernel(const float* __restrict__ x,
                       unsigned char* __restrict__ q,
                       float* __restrict__ st,
                       int M, int K) {
    const int KB   = K >> 7;
    const int lane = threadIdx.x & 31;
    const int gw   = (int)((blockIdx.x * blockDim.x + threadIdx.x) >> 5);
    const int m    = gw / KB;
    const int kb   = gw - m * KB;
    if (m >= M) return;

    const size_t base = (size_t)m * K + (size_t)kb * 128 + lane * 4;
    float4 v = *(const float4*)(x + base);
    float mx = fmaxf(fmaxf(fabsf(v.x), fabsf(v.y)), fmaxf(fabsf(v.z), fabsf(v.w)));
    #pragma unroll
    for (int off = 16; off >= 1; off >>= 1)
        mx = fmaxf(mx, __shfl_xor(mx, off, 32));

    float scale = fmaxf(mx, QEPS) * (1.0f / FP8_MAX_F);
    float inv   = 1.0f / scale;

    unsigned int p = f32_to_fp8_e4m3(v.x * inv)
                   | (f32_to_fp8_e4m3(v.y * inv) << 8)
                   | (f32_to_fp8_e4m3(v.z * inv) << 16)
                   | (f32_to_fp8_e4m3(v.w * inv) << 24);
    *(unsigned int*)(q + base) = p;
    if (lane == 0) st[(size_t)kb * M + m] = scale;
}

// Per (kb, column n): amax over 128 rows, quantize, scatter into the
// WMMA 16x16x128 fp8 B-fragment layout:
//   group = (kb, n/16); within group: lane = (n&15) + 16*h, byte = v*4 + (k&3)
//   h = (k>>3)&1 ; v = (k>>6)*8 + ((k>>4)&3)*2 + ((k>>2)&1)
__global__ __launch_bounds__(256)
void quant_pack_w_kernel(const float* __restrict__ w,   // [K, N] row-major
                         unsigned char* __restrict__ qp,
                         float* __restrict__ sw,        // [KB, N]
                         int K, int N) {
    const int n  = (int)(blockIdx.x * blockDim.x + threadIdx.x);
    const int kb = (int)blockIdx.y;
    if (n >= N) return;

    const float* p = w + (size_t)(kb * 128) * N + n;
    float mx = 0.0f;
    #pragma unroll 8
    for (int r = 0; r < 128; ++r)
        mx = fmaxf(mx, fabsf(p[(size_t)r * N]));

    float scale = fmaxf(mx, QEPS) * (1.0f / FP8_MAX_F);
    float inv   = 1.0f / scale;
    sw[(size_t)kb * N + n] = scale;

    const int lane_lo = n & 15;
    unsigned char* dst = qp + ((size_t)(kb * (N >> 4) + (n >> 4)) * 32) * 64;
    #pragma unroll 8
    for (int k = 0; k < 128; ++k) {
        unsigned int h  = ((unsigned)k >> 3) & 1u;
        unsigned int vv = (((unsigned)k >> 6) << 3)
                        + ((((unsigned)k >> 4) & 3u) << 1)
                        + (((unsigned)k >> 2) & 1u);
        unsigned int b  = (vv << 2) + ((unsigned)k & 3u);
        unsigned int lane = (unsigned)lane_lo + (h << 4);
        dst[lane * 64u + b] = (unsigned char)f32_to_fp8_e4m3(p[(size_t)k * N] * inv);
    }
}

// Issue one 32-byte async global->LDS copy (two b128; inst offset applies to
// both the LDS and the global address, so one address pair suffices).
__device__ __forceinline__ void async_copy32(unsigned lds_off,
                                             const unsigned char* g) {
    asm volatile("global_load_async_to_lds_b128 %0, %1, off\n\t"
                 "global_load_async_to_lds_b128 %0, %1, off offset:16"
                 :: "v"(lds_off), "v"(g)
                 : "memory");
}

__device__ __forceinline__ void wait_async0() {
    asm volatile("s_wait_asynccnt 0x0" ::: "memory");
}

// Block-scale fp8 GEMM. Workgroup = 256 threads = 8 waves; tile 128(M) x 64(N).
// Each wave: 16 rows x 64 cols = 4 accumulators of 16x16. The per-kb B slab
// (8KB, shared by all 8 waves) is double-buffered in LDS via async copies.
// WMMAs are issued in pairs with independent partial destinations so the
// matrix pipe keeps working while the previous pair's scale-FMAs retire.
template <bool GELU>
__global__ __launch_bounds__(256)
void gemm_fp8_blockscale_kernel(const unsigned char* __restrict__ qa, // [M,K] fp8
                                const float* __restrict__ sat,        // [KB, M]
                                const unsigned char* __restrict__ qbp,// packed B
                                const float* __restrict__ sb,         // [KB, N]
                                const float* __restrict__ bias,       // [N]
                                float* __restrict__ out,              // [M, N]
                                int M, int N, int K) {
    const int KB    = K >> 7;
    const int tid   = (int)threadIdx.x;
    const int lane  = tid & 31;
    const int wave  = tid >> 5;
    const int hh    = lane >> 4;     // lane half
    const int ll    = lane & 15;     // low lane id (row for A / col for B,C)
    const int m_base = (int)blockIdx.y * 128 + wave * 16;
    const int n_base = (int)blockIdx.x * 64;

    __shared__ __align__(64) unsigned char lds_b[2][8192];

    union AF { v16i v; unsigned long long u64[8]; };
    union BF { v16i v; v4i q[4]; };

    v8f acc[4];
    #pragma unroll
    for (int nc = 0; nc < 4; ++nc)
        #pragma unroll
        for (int i = 0; i < 8; ++i) acc[nc][i] = 0.0f;

    const unsigned char* arow = qa + (size_t)(m_base + ll) * K + hh * 8;
    const size_t bslab = (size_t)(N >> 4) * 2048;  // qbp bytes per K-block
    // This workgroup's 4 column-groups are contiguous 8KB inside each kb slab.
    const unsigned char* bsrc = qbp + (size_t)(n_base >> 4) * 2048 + tid * 32;
    const unsigned lds_off0 = (unsigned)(size_t)&lds_b[0][tid * 32];
    const unsigned lds_off1 = (unsigned)(size_t)&lds_b[1][tid * 32];

    // prologue: stage kb = 0
    async_copy32(lds_off0, bsrc);

    for (int kb = 0; kb < KB; ++kb) {
        wait_async0();       // my slice of buffer kb has landed in LDS
        __syncthreads();     // everyone's slice landed; prev reads retired

        if (kb + 1 < KB)     // stage next slab into the other buffer
            async_copy32(((kb + 1) & 1) ? lds_off1 : lds_off0,
                         bsrc + (size_t)(kb + 1) * bslab);

        // ---- A fragment: 16x128 fp8, per-ISA layout; 8 contiguous b64 loads ----
        AF a;
        const unsigned char* ap = arow + (size_t)kb * 128;
        #pragma unroll
        for (int t = 0; t < 8; ++t) {
            const int koff = ((t >> 2) << 6) + ((t & 3) << 4); // 0,16,32,48,64,...
            a.u64[t] = *(const unsigned long long*)(ap + koff);
        }
        if (kb + 1 < KB) __builtin_prefetch(ap + 128, 0, 0);

        // ---- per-lane 8 row scales (contiguous in transposed scale buffer) ----
        const float4* sap = (const float4*)(sat + (size_t)kb * M + m_base + hh * 8);
        float4 s0 = sap[0], s1 = sap[1];
        const float sarr[8] = {s0.x, s0.y, s0.z, s0.w, s1.x, s1.y, s1.z, s1.w};

        const float* sbp = sb + (size_t)kb * N + n_base + ll;
        const unsigned char* lbase = &lds_b[kb & 1][lane * 64];

        #pragma unroll
        for (int np = 0; np < 2; ++np) {
            const int nc0 = 2 * np, nc1 = 2 * np + 1;
            BF b0, b1;
            const v4i* bp0 = (const v4i*)(lbase + nc0 * 2048);
            const v4i* bp1 = (const v4i*)(lbase + nc1 * 2048);
            b0.q[0] = bp0[0]; b0.q[1] = bp0[1]; b0.q[2] = bp0[2]; b0.q[3] = bp0[3];
            b1.q[0] = bp1[0]; b1.q[1] = bp1[1]; b1.q[2] = bp1[2]; b1.q[3] = bp1[3];

            const float sb0 = sbp[nc0 * 16];
            const float sb1 = sbp[nc1 * 16];
            v8f zero = {};
            // Two back-to-back WMMAs with independent destinations.
            v8f p0 = __builtin_amdgcn_wmma_f32_16x16x128_fp8_fp8(
                a.v, b0.v, (short)0, zero, false, false);
            v8f p1 = __builtin_amdgcn_wmma_f32_16x16x128_fp8_fp8(
                a.v, b1.v, (short)0, zero, false, false);
            #pragma unroll
            for (int i = 0; i < 8; ++i)
                acc[nc0][i] += sarr[i] * sb0 * p0[i];
            #pragma unroll
            for (int i = 0; i < 8; ++i)
                acc[nc1][i] += sarr[i] * sb1 * p1[i];
        }
    }

    // ---- epilogue: bias (+ optional tanh-GELU), store f32 ----
    #pragma unroll
    for (int nc = 0; nc < 4; ++nc) {
        const int col = n_base + nc * 16 + ll;
        const float bc = bias[col];
        #pragma unroll
        for (int i = 0; i < 8; ++i) {
            const int row = m_base + hh * 8 + i;
            float v = acc[nc][i] + bc;
            if (GELU) {
                const float x3 = v * v * v;
                const float t  = tanhf(0.7978845608028654f * (v + 0.044715f * x3));
                v = 0.5f * v * (1.0f + t);
            }
            out[(size_t)row * N + col] = v;
        }
    }
}

extern "C" void kernel_launch(void* const* d_in, const int* in_sizes, int n_in,
                              void* d_out, int out_size, void* d_ws, size_t ws_size,
                              hipStream_t stream) {
    (void)in_sizes; (void)n_in; (void)out_size; (void)ws_size;

    const float* x  = (const float*)d_in[0];   // [4,2048,2048] -> [8192,2048]
    const float* w1 = (const float*)d_in[1];   // [2048,8192]
    const float* b1 = (const float*)d_in[2];   // [8192]
    const float* w2 = (const float*)d_in[3];   // [8192,8192]
    const float* b2 = (const float*)d_in[4];   // [8192]

    const int M = 8192, K1 = 2048, N1 = 8192, K2 = 8192, N2 = 8192;
    const int KB1 = K1 >> 7, KB2 = K2 >> 7;

    char* ws = (char*)d_ws;
    auto take = [&](size_t bytes) {
        char* p = ws;
        ws += (bytes + 255) & ~(size_t)255;
        return p;
    };
    unsigned char* qx  = (unsigned char*)take((size_t)M  * K1);
    float*         sxt = (float*)        take((size_t)KB1 * M  * 4);
    unsigned char* qw1 = (unsigned char*)take((size_t)K1 * N1);
    float*         sw1 = (float*)        take((size_t)KB1 * N1 * 4);
    unsigned char* qh  = (unsigned char*)take((size_t)M  * N1);
    float*         sht = (float*)        take((size_t)KB2 * M  * 4);
    unsigned char* qw2 = (unsigned char*)take((size_t)K2 * N2);
    float*         sw2 = (float*)        take((size_t)KB2 * N2 * 4);

    // d_out ([8192,8192] f32) doubles as scratch for h; GEMM2 overwrites it.
    float* hbuf = (float*)d_out;

    quant_rows_kernel<<<dim3((M * KB1) / 8), 256, 0, stream>>>(x, qx, sxt, M, K1);
    quant_pack_w_kernel<<<dim3(N1 / 256, KB1), 256, 0, stream>>>(w1, qw1, sw1, K1, N1);
    gemm_fp8_blockscale_kernel<true><<<dim3(N1 / 64, M / 128), 256, 0, stream>>>(
        qx, sxt, qw1, sw1, b1, hbuf, M, N1, K1);

    quant_rows_kernel<<<dim3((M * KB2) / 8), 256, 0, stream>>>(hbuf, qh, sht, M, K2);
    quant_pack_w_kernel<<<dim3(N2 / 256, KB2), 256, 0, stream>>>(w2, qw2, sw2, K2, N2);
    gemm_fp8_blockscale_kernel<false><<<dim3(N2 / 64, M / 128), 256, 0, stream>>>(
        qh, sht, qw2, sw2, b2, (float*)d_out, M, N2, K2);
}
